// ExpertGateNet_20126216749450
// MI455X (gfx1250) — matrix operand order repaired and checked by reading, not compiled, not run on hardware
//
#include <hip/hip_runtime.h>
#include <math.h>

#define HC    128
#define NH    8
#define CPH   16
#define NGRPH 64
#define NEXP  2
#define SLOPE 0.2f

typedef __attribute__((ext_vector_type(16))) _Float16 v16h;
typedef __attribute__((ext_vector_type(8)))  float    v8f;

#define CDIV(a, b) (((a) + (b) - 1) / (b))

// ---------------- helpers ----------------
__device__ __forceinline__ unsigned enc_f32(float f) {
  unsigned u = __float_as_uint(f);
  return (u & 0x80000000u) ? ~u : (u | 0x80000000u);
}
__device__ __forceinline__ float dec_f32(unsigned u) {
  return (u & 0x80000000u) ? __uint_as_float(u & 0x7FFFFFFFu)
                           : __uint_as_float(~u);
}

__global__ void __launch_bounds__(256) fill_f32(float* p, float v, int n) {
  int i = blockIdx.x * 256 + threadIdx.x;
  if (i < n) p[i] = v;
}

// coef[h] = sum_c we[h*16+c] * ae[h*16+c]   (edge attention collapses to a scalar per head)
__global__ void coef_kernel(const float* __restrict__ we, const float* __restrict__ ae,
                            float* __restrict__ coef) {
  int h = threadIdx.x;
  if (h < NH) {
    float s = 0.f;
#pragma unroll
    for (int c = 0; c < CPH; ++c) s += we[h * CPH + c] * ae[h * CPH + c];
    coef[h] = s;
  }
}

// layer-1 linear: h[n,c] = sum_{k<4} x[n,k] * w1[k,c]
__global__ void __launch_bounds__(256) lin1_kernel(const float* __restrict__ x,
                                                   const float* __restrict__ w1,
                                                   float* __restrict__ h, int n) {
  int i = blockIdx.x * 256 + threadIdx.x;
  if (i >= n * HC) return;
  int nd = i >> 7, c = i & 127;
  float s = 0.f;
#pragma unroll
  for (int k = 0; k < 4; ++k) s += x[nd * 4 + k] * w1[k * HC + c];
  h[i] = s;
}

// per-node per-head attention logits: asrc[n,h]=dot(h[n,h,:],as[h,:]), adst likewise
__global__ void __launch_bounds__(256) alpha_nodes(const float* __restrict__ h,
                                                   const float* __restrict__ a_s,
                                                   const float* __restrict__ a_d,
                                                   float* __restrict__ asrc,
                                                   float* __restrict__ adst, int n) {
  int i = blockIdx.x * 256 + threadIdx.x;
  if (i >= n * NH) return;
  int nd = i >> 3, hd = i & 7;
  const float* hp = h + (long)nd * HC + hd * CPH;
  float s0 = 0.f, s1 = 0.f;
#pragma unroll
  for (int c = 0; c < CPH; ++c) {
    float v = hp[c];
    s0 += v * a_s[hd * CPH + c];
    s1 += v * a_d[hd * CPH + c];
  }
  asrc[i] = s0;
  adst[i] = s1;
}

// self-loop 'mean' fill: cnt + sum of incoming edge attrs
__global__ void __launch_bounds__(256) selfloop_accum(const int* __restrict__ dst,
                                                      const float* __restrict__ ea,
                                                      float* cntf, float* loop, int E) {
  int e = blockIdx.x * 256 + threadIdx.x;
  if (e < E) {
    int d = dst[e];
    atomicAdd(&cntf[d], 1.0f);
    atomicAdd(&loop[d], ea[e]);
  }
}
__global__ void __launch_bounds__(256) selfloop_div(float* loop, const float* cntf, int n) {
  int i = blockIdx.x * 256 + threadIdx.x;
  if (i < n) loop[i] /= fmaxf(cntf[i], 1.0f);
}

// pass A: alpha = leaky_relu(asrc[s]+adst[d]+attr*coef); store; atomic segment-max (encoded u32)
__global__ void __launch_bounds__(256) edge_alpha_max(
    const int* __restrict__ src, const int* __restrict__ dst, const float* __restrict__ ea,
    const float* __restrict__ loop, const float* __restrict__ coef,
    const float* __restrict__ asrc, const float* __restrict__ adst,
    float* __restrict__ albuf, unsigned* __restrict__ menc, int E, int nloops) {
  int i = blockIdx.x * 256 + threadIdx.x;
  int tot = (E + nloops) * NH;
  if (i >= tot) return;
  int e = i >> 3, hd = i & 7;
  int s, d;
  float attr;
  if (e < E) { s = src[e]; d = dst[e]; attr = ea[e]; }
  else       { s = d = e - E; attr = loop[e - E]; }
  float a = asrc[s * NH + hd] + adst[d * NH + hd] + attr * coef[hd];
  a = (a > 0.f) ? a : SLOPE * a;
  albuf[i] = a;
  atomicMax(&menc[d * NH + hd], enc_f32(a));
}

// pass B: ex = exp(alpha - m[dst]); atomic segment-sum of denominators
__global__ void __launch_bounds__(256) edge_exp_sum(
    const int* __restrict__ dst, float* __restrict__ albuf,
    const unsigned* __restrict__ menc, float* __restrict__ den, int E, int nloops) {
  int i = blockIdx.x * 256 + threadIdx.x;
  int tot = (E + nloops) * NH;
  if (i >= tot) return;
  int e = i >> 3, hd = i & 7;
  int d = (e < E) ? dst[e] : (e - E);
  float m = dec_f32(menc[d * NH + hd]);
  float ex = __expf(albuf[i] - m);
  albuf[i] = ex;
  atomicAdd(&den[d * NH + hd], ex);
}

// pass C: agg[dst] += h[src] * (ex/den)  -- 32 lanes/edge, float4 per lane, atomics land in L2
__global__ void __launch_bounds__(256) scatter_agg(
    const int* __restrict__ src, const int* __restrict__ dst,
    const float* __restrict__ hlin, const float* __restrict__ albuf,
    const float* __restrict__ den, float* __restrict__ agg, int E, int nloops) {
  long i = (long)blockIdx.x * 256 + threadIdx.x;
  long tot = (long)(E + nloops) * 32;
  if (i >= tot) return;
  int e = (int)(i >> 5);
  int t = (int)(i & 31);
  int hd = t >> 2;
  int s, d;
  if (e < E) { s = src[e]; d = dst[e]; }
  else       { s = d = e - E; }
  float w = albuf[(long)e * NH + hd] / (den[d * NH + hd] + 1e-16f);
  float4 v = *(const float4*)(hlin + (long)s * HC + t * 4);
  float* o = agg + (long)d * HC + t * 4;
  atomicAdd(o + 0, v.x * w);
  atomicAdd(o + 1, v.y * w);
  atomicAdd(o + 2, v.z * w);
  atomicAdd(o + 3, v.w * w);
}

__global__ void __launch_bounds__(256) bias_relu(float* __restrict__ h,
                                                 const float* __restrict__ b, int n) {
  int i = blockIdx.x * 256 + threadIdx.x;
  if (i < n * HC) h[i] = fmaxf(h[i] + b[i & 127], 0.f);
}

// WMMA GEMM: D[M,128] = A[M,128] x W[128,128]; one 16x16 D-tile per wave, 8 waves/block
__global__ void __launch_bounds__(256) gemm128_wmma(const float* __restrict__ A,
                                                    const float* __restrict__ W,
                                                    float* __restrict__ D, int M) {
  __shared__ _Float16 As[16 * 132];  // staged + f16-converted A tile, padded stride
  int mb = blockIdx.x * 16;
  int lane = threadIdx.x & 31;
  int wave = threadIdx.x >> 5;
  bool full = (mb + 16 <= M);  // uniform per block; M%16==0 => always true

  if (full) {
    for (int i = threadIdx.x; i < 16 * HC; i += 256) {
      int r = i >> 7, c = i & 127;
      As[r * 132 + c] = (_Float16)A[(long)(mb + r) * HC + c];
    }
  } else {
    for (int i = threadIdx.x; i < 16 * HC; i += 256) {
      int r = i >> 7, c = i & 127;
      int row = mb + r;
      if (row >= M) row = M - 1;
      As[r * 132 + c] = (_Float16)A[(long)row * HC + c];
    }
  }
  __syncthreads();

  int ncol  = (wave << 4) + (lane & 15);
  int khalf = lane >> 4;
  int arow  = lane & 15;
  v8f acc = {};
#pragma unroll
  for (int kt = 0; kt < 4; ++kt) {
    v16h af, bf;
#pragma unroll
    for (int j = 0; j < 8; ++j) {
      // A 16x32 f16 layout (ISA 7.12.2): lanes 0-15 K=0-7,16-23 ; lanes 16-31 +8
      int ka = ((j & 3) << 1) + ((j >> 2) << 4) + (khalf << 3) + (kt << 5);
      af[2 * j]     = As[arow * 132 + ka];
      af[2 * j + 1] = As[arow * 132 + ka + 1];
      // B 32x16 f16 layout: same K order as A (lanes 0-15 K=0-7,16-23 ; lanes 16-31 +8)
      int kb = ((j & 3) << 1) + ((j >> 2) << 4) + (khalf << 3) + (kt << 5);
      bf[2 * j]     = (_Float16)W[(long)kb * HC + ncol];
      bf[2 * j + 1] = (_Float16)W[(long)(kb + 1) * HC + ncol];
    }
    acc = __builtin_amdgcn_wmma_f32_16x16x32_f16(false, af, false, bf,
                                                 (short)0, acc, false, false);
  }
  // D layout: VGPR r holds row r (+8 for lanes 16-31), col = ncol
  if (full) {
#pragma unroll
    for (int r = 0; r < 8; ++r) {
      int m = mb + r + (khalf << 3);
      D[(long)m * HC + ncol] = acc[r];
    }
  } else {
#pragma unroll
    for (int r = 0; r < 8; ++r) {
      int m = mb + r + (khalf << 3);
      if (m < M) D[(long)m * HC + ncol] = acc[r];
    }
  }
}

// pooling + gate MLP
__global__ void __launch_bounds__(256) pool_accum(const float* __restrict__ h,
                                                  const int* __restrict__ batch,
                                                  float* gsum, float* gcnt, int n) {
  int i = blockIdx.x * 256 + threadIdx.x;
  if (i >= n * HC) return;
  int nd = i >> 7, c = i & 127;
  int g = batch[nd];
  atomicAdd(&gsum[g * HC + c], h[i]);
  if (c == 0) atomicAdd(&gcnt[g], 1.0f);
}
__global__ void pool_div(const float* gsum, const float* gcnt, float* emb) {
  int i = blockIdx.x * 256 + threadIdx.x;
  if (i < NGRPH * HC) emb[i] = gsum[i] / fmaxf(gcnt[i >> 7], 1.0f);
}
__global__ void mlp1_kernel(const float* __restrict__ emb, const float* __restrict__ fw1,
                            const float* __restrict__ fb1, float* __restrict__ gh) {
  int i = blockIdx.x * 256 + threadIdx.x;
  if (i >= NGRPH * 64) return;
  int g = i >> 6, j = i & 63;
  float s = fb1[j];
#pragma unroll 4
  for (int c = 0; c < HC; ++c) s += emb[g * HC + c] * fw1[c * 64 + j];
  gh[i] = fmaxf(s, 0.f);
}
__global__ void mlp2_kernel(const float* __restrict__ gh, const float* __restrict__ fw2,
                            const float* __restrict__ fb2, float* __restrict__ out) {
  int i = blockIdx.x * 256 + threadIdx.x;
  if (i >= NGRPH * NEXP) return;
  int g = i >> 1, k = i & 1;
  float s = fb2[k];
#pragma unroll 4
  for (int j = 0; j < 64; ++j) s += gh[g * 64 + j] * fw2[j * NEXP + k];
  out[i] = s;
}

// ---------------- host orchestration ----------------
extern "C" void kernel_launch(void* const* d_in, const int* in_sizes, int n_in,
                              void* d_out, int out_size, void* d_ws, size_t ws_size,
                              hipStream_t stream) {
  const float* x     = (const float*)d_in[0];
  const int*   ei    = (const int*)d_in[1];
  const float* ea    = (const float*)d_in[2];
  const int*   batch = (const int*)d_in[3];
  const int E = in_sizes[2];
  const int N = in_sizes[3];
  const int* srcI = ei;
  const int* dstI = ei + E;

  const float* W[3]  = {(const float*)d_in[4],  (const float*)d_in[10], (const float*)d_in[16]};
  const float* AS[3] = {(const float*)d_in[5],  (const float*)d_in[11], (const float*)d_in[17]};
  const float* AD[3] = {(const float*)d_in[6],  (const float*)d_in[12], (const float*)d_in[18]};
  const float* WE[3] = {(const float*)d_in[7],  (const float*)d_in[13], (const float*)d_in[19]};
  const float* AE[3] = {(const float*)d_in[8],  (const float*)d_in[14], (const float*)d_in[20]};
  const float* B[3]  = {(const float*)d_in[9],  (const float*)d_in[15], (const float*)d_in[21]};
  const float* fw1 = (const float*)d_in[22];
  const float* fb1 = (const float*)d_in[23];
  const float* fw2 = (const float*)d_in[24];
  const float* fb2 = (const float*)d_in[25];

  // workspace carve-out (256B aligned)
  char* base = (char*)d_ws;
  size_t off = 0;
  auto alloc = [&](size_t bytes) -> void* {
    off = (off + 255) & ~(size_t)255;
    void* p = base + off;
    off += bytes;
    return p;
  };
  float*    hA    = (float*)alloc((size_t)N * HC * 4);        // linear-transform output
  float*    hB    = (float*)alloc((size_t)N * HC * 4);        // layer output / next input
  float*    asrc  = (float*)alloc((size_t)N * NH * 4);
  float*    adst  = (float*)alloc((size_t)N * NH * 4);
  unsigned* menc  = (unsigned*)alloc((size_t)N * NH * 4);
  float*    den   = (float*)alloc((size_t)N * NH * 4);
  float*    albuf = (float*)alloc((size_t)(E + N) * NH * 4);
  float*    loop  = (float*)alloc((size_t)N * 4);
  float*    cntf  = (float*)alloc((size_t)N * 4);
  float*    coef  = (float*)alloc(32 * 4);
  float*    gsum  = (float*)alloc((size_t)NGRPH * HC * 4);
  float*    gcnt  = (float*)alloc((size_t)NGRPH * 4);
  float*    emb   = (float*)alloc((size_t)NGRPH * HC * 4);
  float*    ghb   = (float*)alloc((size_t)NGRPH * 64 * 4);

  const int NC  = N * HC;
  const int NH8 = N * NH;

  // self-loop attrs (mean of incoming) for layers 2/3
  fill_f32<<<CDIV(N, 256), 256, 0, stream>>>(cntf, 0.f, N);
  fill_f32<<<CDIV(N, 256), 256, 0, stream>>>(loop, 0.f, N);
  selfloop_accum<<<CDIV(E, 256), 256, 0, stream>>>(dstI, ea, cntf, loop, E);
  selfloop_div<<<CDIV(N, 256), 256, 0, stream>>>(loop, cntf, N);

  for (int l = 0; l < 3; ++l) {
    int nloops = (l == 0) ? 0 : N;
    int Etot = E + nloops;
    // linear transform -> hA
    if (l == 0)
      lin1_kernel<<<CDIV(NC, 256), 256, 0, stream>>>(x, W[0], hA, N);
    else
      gemm128_wmma<<<CDIV(N, 16), 256, 0, stream>>>(hB, W[l], hA, N);
    // attention logits per node + per-edge scalar coef
    coef_kernel<<<1, 32, 0, stream>>>(WE[l], AE[l], coef);
    alpha_nodes<<<CDIV(NH8, 256), 256, 0, stream>>>(hA, AS[l], AD[l], asrc, adst, N);
    // segment softmax over incoming edges
    fill_f32<<<CDIV(NH8, 256), 256, 0, stream>>>((float*)menc, 0.f, NH8);
    fill_f32<<<CDIV(NH8, 256), 256, 0, stream>>>(den, 0.f, NH8);
    edge_alpha_max<<<CDIV(Etot * NH, 256), 256, 0, stream>>>(
        srcI, dstI, ea, loop, coef, asrc, adst, albuf, menc, E, nloops);
    edge_exp_sum<<<CDIV(Etot * NH, 256), 256, 0, stream>>>(dstI, albuf, menc, den, E, nloops);
    // weighted scatter-aggregate into hB, then bias+relu
    fill_f32<<<CDIV(NC, 256), 256, 0, stream>>>(hB, 0.f, NC);
    long sc = (long)Etot * 32;
    scatter_agg<<<(int)CDIV(sc, 256), 256, 0, stream>>>(srcI, dstI, hA, albuf, den, hB, E, nloops);
    bias_relu<<<CDIV(NC, 256), 256, 0, stream>>>(hB, B[l], N);
  }

  // global mean pool + gate MLP
  fill_f32<<<CDIV(NGRPH * HC, 256), 256, 0, stream>>>(gsum, 0.f, NGRPH * HC);
  fill_f32<<<1, 256, 0, stream>>>(gcnt, 0.f, NGRPH);
  pool_accum<<<CDIV(NC, 256), 256, 0, stream>>>(hB, batch, gsum, gcnt, N);
  pool_div<<<CDIV(NGRPH * HC, 256), 256, 0, stream>>>(gsum, gcnt, emb);
  mlp1_kernel<<<CDIV(NGRPH * 64, 256), 256, 0, stream>>>(emb, fw1, fb1, ghb);
  mlp2_kernel<<<1, 256, 0, stream>>>(ghb, fw2, fb2, (float*)d_out);
}
